// SelfAttention_17343077941592
// MI455X (gfx1250) — matrix operand order repaired and checked by reading, not compiled, not run on hardware
//
#include <hip/hip_runtime.h>
#include <hip/hip_bf16.h>

// ---------------------------------------------------------------------------
// Self-attention block (GroupNorm -> qkv 1x1 -> softmax(q^T k) v -> proj + res)
// B=4, C=512, H=W=64 (HW=4096), 32 groups. fp16 WMMA w/ fp32 accumulate,
// two-pass flash softmax (no 268MB score materialization).
// ---------------------------------------------------------------------------

#define Bn      4
#define Cn      512
#define HWn     4096
#define GROUPSn 32
#define OC3     1536
#define CPG     (Cn / GROUPSn)          // 16
#define GELEMS  (CPG * HWn)             // 65536 per (b,g)

static __device__ __constant__ float kScale = 0.04419417382415922f; // 1/sqrt(512)

typedef __attribute__((ext_vector_type(16))) _Float16 v16h;
typedef __attribute__((ext_vector_type(8)))  float    v8f;

// ---------------- WMMA helpers (CDNA5 16x16x32 f16 -> f32) -----------------

__device__ inline v8f wmma16(v16h a, v16h b, v8f c) {
  // (neg_a, A, neg_b, B, c_mod, C, reuse_a, reuse_b)
  return __builtin_amdgcn_wmma_f32_16x16x32_f16(false, a, false, b, (short)0, c,
                                                false, false);
}

// A fragment: 16x32 (MxK) f16, source row-major [m][k] with row stride ld.
// ISA layout: lane m=l&15, half=l>>4; vreg r holds K = (r&4?16:0)+8*half+2*(r&3).
// K pairs are contiguous -> compiler can merge into 32-bit loads.
__device__ inline v16h load_a_frag(const _Float16* base, int ld) {
  const int l = threadIdx.x & 31;
  const int m = l & 15;
  const int hf = l >> 4;
  v16h a;
#pragma unroll
  for (int r = 0; r < 8; ++r) {
    const int k = ((r & 4) << 2) + (hf << 3) + ((r & 3) << 1);
    a[2 * r]     = base[m * ld + k];
    a[2 * r + 1] = base[m * ld + k + 1];
  }
  return a;
}

// B fragment: 32x16 (KxN) f16 where B is stored TRANSPOSED: Bt[n][k], row
// stride ld.  ISA layout (per SWMMAC B tables): lane n=l&15, half=l>>4;
// vreg r holds K = (r&4?16:0)+8*half+2*(r&3) (pairs contiguous in Bt rows).
__device__ inline v16h load_bT_frag(const _Float16* baseT, long ld) {
  const int l = threadIdx.x & 31;
  const int n = l & 15;
  const int hk = (l >> 4) << 3;
  v16h b;
#pragma unroll
  for (int r = 0; r < 8; ++r) {
    const int k = ((r & 4) << 2) + hk + ((r & 3) << 1);
    b[2 * r]     = baseT[(long)n * ld + k];
    b[2 * r + 1] = baseT[(long)n * ld + k + 1];
  }
  return b;
}

// ---------------- kernel 0: weight fp32 -> fp16 ----------------------------

__global__ void cvt_weights(const float* __restrict__ qkvw,
                            const float* __restrict__ projw,
                            _Float16* __restrict__ qkvw16,
                            _Float16* __restrict__ projw16) {
  const int i = blockIdx.x * 256 + threadIdx.x;
  if (i < OC3 * Cn)  qkvw16[i] = (_Float16)qkvw[i];
  if (i < Cn * Cn)   projw16[i] = (_Float16)projw[i];
}

// ---------------- kernel 1: GroupNorm -> fp16 xn ---------------------------

__global__ void __launch_bounds__(256) groupnorm_f16(
    const float* __restrict__ x, const float* __restrict__ gw,
    const float* __restrict__ gb, _Float16* __restrict__ xn) {
  const int bg = blockIdx.x;
  const int b = bg >> 5, g = bg & 31;
  const long base = (long)b * Cn * HWn + (long)g * CPG * HWn;
  const int tid = threadIdx.x;

  float s = 0.f, s2 = 0.f;
  for (int idx = tid; idx < GELEMS; idx += 256) {
    const float v = x[base + idx];
    s += v; s2 += v * v;
  }
  __shared__ float r0[256], r1[256];
  r0[tid] = s; r1[tid] = s2;
  __syncthreads();
#pragma unroll
  for (int st = 128; st > 0; st >>= 1) {
    if (tid < st) { r0[tid] += r0[tid + st]; r1[tid] += r1[tid + st]; }
    __syncthreads();
  }
  const float inv_n = 1.0f / (float)GELEMS;
  const float mean = r0[0] * inv_n;
  const float var  = r1[0] * inv_n - mean * mean;
  const float rstd = rsqrtf(var + 1e-6f);

  for (int idx = tid; idx < GELEMS; idx += 256) {
    const int c = g * CPG + (idx >> 12);        // idx / HWn
    const float v = (x[base + idx] - mean) * rstd * gw[c] + gb[c];
    xn[base + idx] = (_Float16)v;
  }
}

// ---------------- kernel 2: QKV GEMM (1536x4096 = W[1536x512] @ xn) --------
// Block tile 128(M) x 64(N), 8 waves as 4x2, each wave 2x2 WMMA tiles.
// Outputs: qT[b][i][c], kT[b][j][c] (both transposed), v[b][c][j].

__global__ void __launch_bounds__(256) qkv_gemm(
    const _Float16* __restrict__ W, const float* __restrict__ bias,
    const _Float16* __restrict__ xn, _Float16* __restrict__ qT,
    _Float16* __restrict__ kT, _Float16* __restrict__ v) {
  const int b  = blockIdx.z;
  const int m0 = blockIdx.y * 128;
  const int n0 = blockIdx.x * 64;
  const int tid = threadIdx.x;
  const int w = tid >> 5, wm = w >> 1, wn = w & 1;

  __shared__ _Float16 As[128][34];     // [m][k]
  __shared__ _Float16 Bt[64][34];      // [n][k]  (transposed stage)

  const _Float16* xb = xn + (long)b * Cn * HWn;
  v8f acc[2][2] = {};

  for (int kk = 0; kk < Cn; kk += 32) {
    for (int idx = tid; idx < 128 * 32; idx += 256) {
      const int r = idx >> 5, c = idx & 31;
      As[r][c] = W[(long)(m0 + r) * Cn + kk + c];
    }
    for (int idx = tid; idx < 32 * 64; idx += 256) {
      const int r = idx >> 6, c = idx & 63;  // r = k row, c = n col
      Bt[c][r] = xb[(long)(kk + r) * HWn + n0 + c];
    }
    __syncthreads();

    v16h a0 = load_a_frag(&As[wm * 32][0], 34);
    v16h a1 = load_a_frag(&As[wm * 32 + 16][0], 34);
    v16h b0 = load_bT_frag(&Bt[wn * 32][0], 34);
    v16h b1 = load_bT_frag(&Bt[wn * 32 + 16][0], 34);
    acc[0][0] = wmma16(a0, b0, acc[0][0]);
    acc[0][1] = wmma16(a0, b1, acc[0][1]);
    acc[1][0] = wmma16(a1, b0, acc[1][0]);
    acc[1][1] = wmma16(a1, b1, acc[1][1]);
    __syncthreads();
  }

  const int l = tid & 31, nl = l & 15, hs = l >> 4;
#pragma unroll
  for (int mt = 0; mt < 2; ++mt)
#pragma unroll
    for (int nt = 0; nt < 2; ++nt)
#pragma unroll
      for (int r = 0; r < 8; ++r) {
        const int m = m0 + wm * 32 + mt * 16 + r + 8 * hs;
        const int n = n0 + wn * 32 + nt * 16 + nl;
        const float val = acc[mt][nt][r] + bias[m];
        if (m0 < Cn) {                 // q -> qT[b][i][c]
          qT[((long)b * HWn + n) * Cn + m] = (_Float16)val;
        } else if (m0 < 2 * Cn) {      // k -> kT[b][j][c]
          kT[((long)b * HWn + n) * Cn + (m - Cn)] = (_Float16)val;
        } else {                       // v -> v[b][c][j]
          v[((long)b * Cn + (m - 2 * Cn)) * HWn + n] = (_Float16)val;
        }
      }
}

// ---------------- kernel 3: pass 1 — row max & sum-exp ---------------------
// Block = 16 queries x 256 threads. Channels split across 8 waves (64 each),
// partial 16x32 score tiles reduced through LDS, per-thread online (m,z).

__global__ void __launch_bounds__(256) attn_stats(
    const _Float16* __restrict__ qT, const _Float16* __restrict__ kT,
    float* __restrict__ rowmax, float* __restrict__ rowsum) {
  const int b  = blockIdx.y;
  const int i0 = blockIdx.x * 16;
  const int tid = threadIdx.x;
  const int w = tid >> 5;

  __shared__ _Float16 Qs[16][520];
  __shared__ float Sred[8][16][33];
  __shared__ float Ms[16][16], Zs[16][16];

  for (int idx = tid; idx < 16 * Cn; idx += 256) {
    const int r = idx >> 9, c = idx & 511;
    Qs[r][c] = qT[((long)b * HWn + i0 + r) * Cn + c];
  }
  __syncthreads();

  const _Float16* kb = kT + (long)b * HWn * Cn;
  const int cw = w * 64;
  const int rr = tid & 15, cg = tid >> 4;
  float mloc = -3.0e38f, zloc = 0.f;

  for (int j0 = 0; j0 < HWn; j0 += 32) {
    v8f s0 = {}, s1 = {};
#pragma unroll
    for (int ks = 0; ks < 2; ++ks) {
      const int cb = cw + ks * 32;
      v16h a  = load_a_frag(&Qs[0][cb], 520);
      v16h bk0 = load_bT_frag(kb + (long)j0 * Cn + cb, Cn);
      v16h bk1 = load_bT_frag(kb + (long)(j0 + 16) * Cn + cb, Cn);
      s0 = wmma16(a, bk0, s0);
      s1 = wmma16(a, bk1, s1);
    }
    const int l = tid & 31, nl = l & 15, hs = l >> 4;
#pragma unroll
    for (int r = 0; r < 8; ++r) {
      Sred[w][r + 8 * hs][nl]      = s0[r];
      Sred[w][r + 8 * hs][16 + nl] = s1[r];
    }
    __syncthreads();
#pragma unroll
    for (int cc = 0; cc < 2; ++cc) {
      const int col = cg * 2 + cc;
      float s = 0.f;
#pragma unroll
      for (int ww = 0; ww < 8; ++ww) s += Sred[ww][rr][col];
      s *= kScale;
      const float mn = fmaxf(mloc, s);
      zloc = zloc * __expf(mloc - mn) + __expf(s - mn);
      mloc = mn;
    }
    __syncthreads();
  }

  Ms[rr][cg] = mloc; Zs[rr][cg] = zloc;
  __syncthreads();
  if (tid < 16) {
    float m = Ms[tid][0], z = Zs[tid][0];
#pragma unroll
    for (int g2 = 1; g2 < 16; ++g2) {
      const float m2 = Ms[tid][g2], z2 = Zs[tid][g2];
      const float mn = fmaxf(m, m2);
      z = z * __expf(m - mn) + z2 * __expf(m2 - mn);
      m = mn;
    }
    rowmax[(long)b * HWn + i0 + tid] = m;
    rowsum[(long)b * HWn + i0 + tid] = z;
  }
}

// ---------------- kernel 4: pass 2 — P = softmax tile, O += P @ V^T --------
// Same score pipeline; fold exp(s-m)/Z into f16 P tile in LDS, each wave
// owns 64 output channels (4 v8f accumulators).

__global__ void __launch_bounds__(256) attn_av(
    const _Float16* __restrict__ qT, const _Float16* __restrict__ kT,
    const _Float16* __restrict__ vmat, const float* __restrict__ rowmax,
    const float* __restrict__ rowsum, _Float16* __restrict__ hmap) {
  const int b  = blockIdx.y;
  const int i0 = blockIdx.x * 16;
  const int tid = threadIdx.x;
  const int w = tid >> 5;

  __shared__ _Float16 Qs[16][520];
  __shared__ float Sred[8][16][33];
  __shared__ _Float16 P[16][34];
  __shared__ float rmax_s[16], rinv_s[16];

  for (int idx = tid; idx < 16 * Cn; idx += 256) {
    const int r = idx >> 9, c = idx & 511;
    Qs[r][c] = qT[((long)b * HWn + i0 + r) * Cn + c];
  }
  if (tid < 16) {
    rmax_s[tid] = rowmax[(long)b * HWn + i0 + tid];
    rinv_s[tid] = 1.0f / rowsum[(long)b * HWn + i0 + tid];
  }
  __syncthreads();

  const _Float16* kb = kT + (long)b * HWn * Cn;
  const _Float16* vb = vmat + (long)b * Cn * HWn;
  const int cw = w * 64;
  const int l = tid & 31, nl = l & 15, hs = l >> 4;

  v8f o_acc[4] = {};

  for (int j0 = 0; j0 < HWn; j0 += 32) {
    // --- scores (channels split across waves) ---
    v8f s0 = {}, s1 = {};
#pragma unroll
    for (int ks = 0; ks < 2; ++ks) {
      const int cb = cw + ks * 32;
      v16h a  = load_a_frag(&Qs[0][cb], 520);
      v16h bk0 = load_bT_frag(kb + (long)j0 * Cn + cb, Cn);
      v16h bk1 = load_bT_frag(kb + (long)(j0 + 16) * Cn + cb, Cn);
      s0 = wmma16(a, bk0, s0);
      s1 = wmma16(a, bk1, s1);
    }
#pragma unroll
    for (int r = 0; r < 8; ++r) {
      Sred[w][r + 8 * hs][nl]      = s0[r];
      Sred[w][r + 8 * hs][16 + nl] = s1[r];
    }
    __syncthreads();

    // --- normalized probability tile (f16) ---
    for (int e = tid; e < 16 * 32; e += 256) {
      const int r = e >> 5, col = e & 31;
      float s = 0.f;
#pragma unroll
      for (int ww = 0; ww < 8; ++ww) s += Sred[ww][r][col];
      s *= kScale;
      P[r][col] = (_Float16)(__expf(s - rmax_s[r]) * rinv_s[r]);
    }
    __syncthreads();

    // --- O(16 x 64 per wave) += P(16x32) @ V^T(32 x 64) ---
    v16h pa = load_a_frag(&P[0][0], 34);
#pragma unroll
    for (int nt = 0; nt < 4; ++nt) {
      // B[k=j][n=c] = v[c][j]; v stored [c][j] -> transposed gather, ld=HWn
      v16h bv = load_bT_frag(vb + (long)(cw + nt * 16) * HWn + j0, HWn);
      o_acc[nt] = wmma16(pa, bv, o_acc[nt]);
    }
    __syncthreads();
  }

  // write hmap[b][c][i] (f16)
#pragma unroll
  for (int nt = 0; nt < 4; ++nt)
#pragma unroll
    for (int r = 0; r < 8; ++r) {
      const int c = cw + nt * 16 + nl;
      const int i = i0 + r + 8 * hs;
      hmap[((long)b * Cn + c) * HWn + i] = (_Float16)o_acc[nt][r];
    }
}

// ---------------- kernel 5: proj GEMM + bias + residual --------------------

__global__ void __launch_bounds__(256) proj_gemm(
    const _Float16* __restrict__ W, const float* __restrict__ bias,
    const _Float16* __restrict__ hmap, const float* __restrict__ x,
    float* __restrict__ out) {
  const int b  = blockIdx.z;
  const int m0 = blockIdx.y * 128;
  const int n0 = blockIdx.x * 64;
  const int tid = threadIdx.x;
  const int w = tid >> 5, wm = w >> 1, wn = w & 1;

  __shared__ _Float16 As[128][34];
  __shared__ _Float16 Bt[64][34];

  const _Float16* hb = hmap + (long)b * Cn * HWn;
  v8f acc[2][2] = {};

  for (int kk = 0; kk < Cn; kk += 32) {
    for (int idx = tid; idx < 128 * 32; idx += 256) {
      const int r = idx >> 5, c = idx & 31;
      As[r][c] = W[(long)(m0 + r) * Cn + kk + c];
    }
    for (int idx = tid; idx < 32 * 64; idx += 256) {
      const int r = idx >> 6, c = idx & 63;
      Bt[c][r] = hb[(long)(kk + r) * HWn + n0 + c];
    }
    __syncthreads();

    v16h a0 = load_a_frag(&As[wm * 32][0], 34);
    v16h a1 = load_a_frag(&As[wm * 32 + 16][0], 34);
    v16h b0 = load_bT_frag(&Bt[wn * 32][0], 34);
    v16h b1 = load_bT_frag(&Bt[wn * 32 + 16][0], 34);
    acc[0][0] = wmma16(a0, b0, acc[0][0]);
    acc[0][1] = wmma16(a0, b1, acc[0][1]);
    acc[1][0] = wmma16(a1, b0, acc[1][0]);
    acc[1][1] = wmma16(a1, b1, acc[1][1]);
    __syncthreads();
  }

  const int l = tid & 31, nl = l & 15, hs = l >> 4;
#pragma unroll
  for (int mt = 0; mt < 2; ++mt)
#pragma unroll
    for (int nt = 0; nt < 2; ++nt)
#pragma unroll
      for (int r = 0; r < 8; ++r) {
        const int m = m0 + wm * 32 + mt * 16 + r + 8 * hs;
        const int n = n0 + wn * 32 + nt * 16 + nl;
        const long o = ((long)b * Cn + m) * HWn + n;
        out[o] = x[o] + acc[mt][nt][r] + bias[m];
      }
}

// ---------------------------------------------------------------------------

extern "C" void kernel_launch(void* const* d_in, const int* in_sizes, int n_in,
                              void* d_out, int out_size, void* d_ws,
                              size_t ws_size, hipStream_t stream) {
  const float* x      = (const float*)d_in[0];
  const float* norm_w = (const float*)d_in[1];
  const float* norm_b = (const float*)d_in[2];
  const float* qkv_w  = (const float*)d_in[3];
  const float* qkv_b  = (const float*)d_in[4];
  const float* proj_w = (const float*)d_in[5];
  const float* proj_b = (const float*)d_in[6];
  float* out = (float*)d_out;

  char* ws = (char*)d_ws;
  size_t off = 0;
  auto alloc = [&](size_t bytes) -> void* {
    void* p = (void*)(ws + off);
    off += (bytes + 255) & ~(size_t)255;
    return p;
  };
  _Float16* qkvw16 = (_Float16*)alloc((size_t)OC3 * Cn * 2);
  _Float16* projw16 = (_Float16*)alloc((size_t)Cn * Cn * 2);
  _Float16* xn   = (_Float16*)alloc((size_t)Bn * Cn * HWn * 2);
  _Float16* qT   = (_Float16*)alloc((size_t)Bn * HWn * Cn * 2);
  _Float16* kT   = (_Float16*)alloc((size_t)Bn * HWn * Cn * 2);
  _Float16* vm   = (_Float16*)alloc((size_t)Bn * Cn * HWn * 2);
  _Float16* hmap = (_Float16*)alloc((size_t)Bn * Cn * HWn * 2);
  float* rowmax  = (float*)alloc((size_t)Bn * HWn * 4);
  float* rowsum  = (float*)alloc((size_t)Bn * HWn * 4);
  (void)ws_size; (void)in_sizes; (void)n_in; (void)out_size;

  // 0) weights to fp16
  cvt_weights<<<(OC3 * Cn + 255) / 256, 256, 0, stream>>>(qkv_w, proj_w,
                                                          qkvw16, projw16);
  // 1) GroupNorm -> fp16 xn
  groupnorm_f16<<<Bn * GROUPSn, 256, 0, stream>>>(x, norm_w, norm_b, xn);
  // 2) QKV GEMM
  qkv_gemm<<<dim3(HWn / 64, OC3 / 128, Bn), 256, 0, stream>>>(
      qkvw16, qkv_b, xn, qT, kT, vm);
  // 3) softmax stats (row max / sum-exp), streaming over keys
  attn_stats<<<dim3(HWn / 16, Bn), 256, 0, stream>>>(qT, kT, rowmax, rowsum);
  // 4) attention output (recompute scores, P @ V^T)
  attn_av<<<dim3(HWn / 16, Bn), 256, 0, stream>>>(qT, kT, vm, rowmax, rowsum,
                                                  hmap);
  // 5) projection + bias + residual
  proj_gemm<<<dim3(HWn / 64, Cn / 128, Bn), 256, 0, stream>>>(
      projw16, proj_b, hmap, x, out);
}
